// TLC_DQN_50852412784681
// MI455X (gfx1250) — matrix operand order repaired and checked by reading, not compiled, not run on hardware
//
#include <hip/hip_runtime.h>
#include <hip/hip_bf16.h>

typedef _Float16 v16h __attribute__((ext_vector_type(16)));
typedef _Float16 h8_t  __attribute__((ext_vector_type(8)));
typedef float    v8f  __attribute__((ext_vector_type(8)));

#define B_   16
#define N_   1000
#define BN_  16000
#define T_   16
#define FO_  16
#define H_   128
#define G3_  384
#define HEADS_ 5
#define HC_  640          // heads*H
#define E_   4000
#define BE_  64000        // B*E
#define ETOT_ 80000       // B*E + BN
#define ROWS_ 64          // rows per GRU block

// ---------------------------------------------------------------- helpers
__device__ __forceinline__ unsigned enc_f(float f) {
    unsigned u = __float_as_uint(f);
    return (u & 0x80000000u) ? ~u : (u | 0x80000000u);
}
__device__ __forceinline__ float dec_f(unsigned u) {
    return __uint_as_float((u & 0x80000000u) ? ~u : (u & 0x7fffffffu));
}
__device__ __forceinline__ void edge_sd(int e, const int* edges, int& s, int& d) {
    if (e < BE_) {
        int b = e / E_, j = e - b * E_;
        s = edges[b * 2 * E_ + j]       + b * N_;
        d = edges[b * 2 * E_ + E_ + j]  + b * N_;
    } else { s = d = e - BE_; }
}
__device__ __forceinline__ float sigmoidf_(float v) {
    return 1.f / (1.f + expf(-v));
}

// ---------------------------------------------------------------- zero ws
__global__ void k_zero(float* p, int n) {
    int i = blockIdx.x * blockDim.x + threadIdx.x;
    if (i < n) p[i] = 0.f;
}

// ------------------------------------------------------ weight conversion
// Wih16/Whh16: [384][128] f16 (n-major, k-contiguous already).
// WlT/WrT:     [640][128] f16 = transpose of Wl/Wr [128][640].
__global__ void k_convert(const float* Wih, const float* Whh,
                          const float* Wl, const float* Wr,
                          _Float16* Wih16, _Float16* Whh16,
                          _Float16* WlT, _Float16* WrT) {
    int id = blockIdx.x * blockDim.x + threadIdx.x;
    const int S1 = G3_ * H_;          // 49152
    const int S2 = HC_ * H_;          // 81920
    if (id < S1)                 { Wih16[id] = (_Float16)Wih[id]; return; }
    id -= S1;
    if (id < S1)                 { Whh16[id] = (_Float16)Whh[id]; return; }
    id -= S1;
    if (id < S2) { int n = id >> 7, k = id & 127; WlT[id] = (_Float16)Wl[k * HC_ + n]; return; }
    id -= S2;
    if (id < S2) { int n = id >> 7, k = id & 127; WrT[id] = (_Float16)Wr[k * HC_ + n]; }
}

// ---------------------------------------------------------------- GRU
// One block = 64 rows, persistent over T=16 steps.
// LDS = xs(f16 64x128) + h double-buffer (2 x f16 64x128) = 48 KB.
// Each wave owns 4 (mtile, h-coltile) groups; for each it accumulates the
// r, z, xn, hn 16x16 tiles in registers (same C-fragment lane mapping), so
// gate math is pure per-lane register arithmetic -- no gate LDS buffer.
__global__ __launch_bounds__(256) void k_gru(
        const float* __restrict__ obs, const float* __restrict__ W_nbrs,
        const float* __restrict__ b_nbrs,
        const float* __restrict__ b_ih, const float* __restrict__ b_hh,
        const _Float16* __restrict__ Wih16, const _Float16* __restrict__ Whh16,
        _Float16* __restrict__ h_out) {
    extern __shared__ char smem[];
    _Float16* xs = (_Float16*)smem;                 // [64][128]
    _Float16* h0 = xs + ROWS_ * H_;                 // [64][128]
    _Float16* h1 = h0 + ROWS_ * H_;                 // [64][128]

    const int tid  = threadIdx.x;
    const int lane = tid & 31;
    const int wave = tid >> 5;
    const int row0 = blockIdx.x * ROWS_;

    for (int i = tid; i < ROWS_ * H_; i += 256) h0[i] = (_Float16)0.f;

    const int erow = tid >> 2;              // 0..63 : row in embed phase
    const int ec0  = (tid & 3) * 32;        // 32-col chunk

    const int lm = lane & 15;
    const int lh = lane >> 4;
    const int aoff = lh * 16;               // A-fragment K sub-offset
    const int bko  = lh * 16;               // B-fragment K sub-offset
    const int mtile = wave >> 1;            // 0..3
    const int ntb   = (wave & 1) * 4;       // h-coltile base (0 or 4)
    const int am    = mtile * 16 + lm;      // A row within tile

    for (int t = 0; t < T_; ++t) {
        __syncthreads();    // prev-step h writes visible; xs free to overwrite

        // ---- per-node obs embedding: x = relu(obs @ W_nbrs + b)
        {
            const float* orow = obs + (size_t)(row0 + erow) * (T_ * FO_) + t * FO_;
            float ov[FO_];
#pragma unroll
            for (int f = 0; f < FO_; ++f) ov[f] = orow[f];
            for (int c = ec0; c < ec0 + 32; ++c) {
                float acc = b_nbrs[c];
#pragma unroll
                for (int f = 0; f < FO_; ++f) acc += ov[f] * W_nbrs[f * H_ + c];
                xs[erow * H_ + c] = (_Float16)fmaxf(acc, 0.f);
            }
        }
        __syncthreads();

        _Float16* hc = (t & 1) ? h1 : h0;   // read
        _Float16* hn = (t & 1) ? h0 : h1;   // write

        for (int gi = 0; gi < 4; ++gi) {
            const int nt  = ntb + gi;
            const int col = nt * 16 + lm;   // output h-column of this lane
            v8f cr = {}, cz = {}, cxn = {}, chn = {};
            const size_t roR = (size_t)(nt * 16 + lm) * H_ + bko;
            const size_t roZ = (size_t)(128 + nt * 16 + lm) * H_ + bko;
            const size_t roN = (size_t)(256 + nt * 16 + lm) * H_ + bko;
            for (int ks = 0; ks < 4; ++ks) {
                const int kb = ks * 32;
                v16h ax, ah, b;
                const _Float16* axp = xs + am * H_ + kb + aoff;
                *(h8_t*)&ax       = *(const h8_t*)axp;
                *((h8_t*)&ax + 1) = *(const h8_t*)(axp + 8);
                const _Float16* ahp = hc + am * H_ + kb + aoff;
                *(h8_t*)&ah       = *(const h8_t*)ahp;
                *((h8_t*)&ah + 1) = *(const h8_t*)(ahp + 8);
                // one A pair feeds 6 WMMAs (r,z,n columns of both weight mats)
                b = *(const v16h*)(Wih16 + roR + kb);
                cr = __builtin_amdgcn_wmma_f32_16x16x32_f16(false, ax, false, b, (short)0, cr, false, false);
                b = *(const v16h*)(Whh16 + roR + kb);
                cr = __builtin_amdgcn_wmma_f32_16x16x32_f16(false, ah, false, b, (short)0, cr, false, false);
                b = *(const v16h*)(Wih16 + roZ + kb);
                cz = __builtin_amdgcn_wmma_f32_16x16x32_f16(false, ax, false, b, (short)0, cz, false, false);
                b = *(const v16h*)(Whh16 + roZ + kb);
                cz = __builtin_amdgcn_wmma_f32_16x16x32_f16(false, ah, false, b, (short)0, cz, false, false);
                b = *(const v16h*)(Wih16 + roN + kb);
                cxn = __builtin_amdgcn_wmma_f32_16x16x32_f16(false, ax, false, b, (short)0, cxn, false, false);
                b = *(const v16h*)(Whh16 + roN + kb);
                chn = __builtin_amdgcn_wmma_f32_16x16x32_f16(false, ah, false, b, (short)0, chn, false, false);
            }
            // ---- in-register gate math (all 4 tiles share lane mapping)
            const float br = b_ih[col] + b_hh[col];
            const float bz = b_ih[128 + col] + b_hh[128 + col];
            const float bxn = b_ih[256 + col];
            const float bhn = b_hh[256 + col];
#pragma unroll
            for (int r = 0; r < 8; ++r) {
                const int row = mtile * 16 + lh * 8 + r;
                float rr = sigmoidf_(cr[r] + br);
                float zz = sigmoidf_(cz[r] + bz);
                float nn = tanhf(cxn[r] + bxn + rr * (chn[r] + bhn));
                float hold = (float)hc[row * H_ + col];
                hn[row * H_ + col] = (_Float16)((1.f - zz) * nn + zz * hold);
            }
        }
    }
    __syncthreads();
    // T_=16 even -> final state sits in h0
    for (int i = tid; i < ROWS_ * H_; i += 256)
        h_out[(size_t)row0 * H_ + i] = h0[i];
}

// ---------------------------------------------------------------- GAT GEMM
// xl = h@Wl, xr = h@Wr   ([16000x128]x[128x640], f16 out)
__global__ __launch_bounds__(256) void k_gat_gemm(
        const _Float16* __restrict__ h,
        const _Float16* __restrict__ WlT, const _Float16* __restrict__ WrT,
        _Float16* __restrict__ xl, _Float16* __restrict__ xr) {
    const int tid = threadIdx.x, lane = tid & 31, wave = tid >> 5;
    const int mbase = blockIdx.x * 16;
    const int arow = mbase + (lane & 15);
    const int aoff = (lane >> 4) * 16;
    const int bko  = (lane >> 4) * 16;
    for (int jj = 0; jj < 10; ++jj) {
        const int job = wave * 10 + jj;                    // 0..79
        const _Float16* Wt = (job < 40) ? WlT : WrT;
        _Float16* outp     = (job < 40) ? xl  : xr;
        const int nt = (job < 40) ? job : job - 40;
        const int nb = nt * 16;
        const int bn = nb + (lane & 15);
        v8f c = {};
        for (int ks = 0; ks < 4; ++ks) {
            const int kb = ks * 32;
            v16h a, b;
            const _Float16* ap = h + (size_t)arow * H_ + kb + aoff;
            *(h8_t*)&a       = *(const h8_t*)ap;
            *((h8_t*)&a + 1) = *(const h8_t*)(ap + 8);
            b = *(const v16h*)(Wt + (size_t)bn * H_ + kb + bko);
            c = __builtin_amdgcn_wmma_f32_16x16x32_f16(
                    false, a, false, b, (short)0, c, false, false);
        }
        const int col   = nb + (lane & 15);
        const int rbase = mbase + (lane >> 4) * 8;
#pragma unroll
        for (int r = 0; r < 8; ++r)
            outp[(size_t)(rbase + r) * HC_ + col] = (_Float16)c[r];
    }
}

// ---------------------------------------------------------------- edges
__global__ void k_edge_logits(const _Float16* __restrict__ xl,
                              const _Float16* __restrict__ xr,
                              const float* __restrict__ att_a,
                              const int* __restrict__ edges,
                              float* __restrict__ logits,
                              unsigned* __restrict__ segmax) {
    int id = blockIdx.x * blockDim.x + threadIdx.x;
    if (id >= ETOT_ * HEADS_) return;
    const int e = id / HEADS_, hh = id - e * HEADS_;
    int s, d; edge_sd(e, edges, s, d);
    const _Float16* pl = xl + (size_t)s * HC_ + hh * H_;
    const _Float16* pr = xr + (size_t)d * HC_ + hh * H_;
    const float* aa = att_a + hh * H_;
    float acc = 0.f;
    for (int c0 = 0; c0 < H_; c0 += 8) {           // 16B vector gathers
        h8_t vl = *(const h8_t*)(pl + c0);
        h8_t vr = *(const h8_t*)(pr + c0);
#pragma unroll
        for (int j = 0; j < 8; ++j) {
            float v = (float)vl[j] + (float)vr[j];
            v = (v > 0.f) ? v : 0.2f * v;          // leaky_relu(0.2)
            acc += v * aa[c0 + j];
        }
    }
    logits[id] = acc;
    atomicMax(&segmax[d * HEADS_ + hh], enc_f(acc));
}

__global__ void k_edge_exp(const int* __restrict__ edges,
                           float* __restrict__ logits,
                           const unsigned* __restrict__ segmax,
                           float* __restrict__ segsum) {
    int id = blockIdx.x * blockDim.x + threadIdx.x;
    if (id >= ETOT_ * HEADS_) return;
    const int e = id / HEADS_, hh = id - e * HEADS_;
    int s, d; edge_sd(e, edges, s, d);
    float p = expf(logits[id] - dec_f(segmax[d * HEADS_ + hh]));
    logits[id] = p;
    atomicAdd(&segsum[d * HEADS_ + hh], p);
}

__global__ void k_edge_alpha(const int* __restrict__ edges,
                             const float* __restrict__ logits,
                             const float* __restrict__ segsum,
                             const _Float16* __restrict__ xl,
                             float* __restrict__ alpha_out,
                             float* __restrict__ node_acc) {
    int id = blockIdx.x * blockDim.x + threadIdx.x;
    if (id >= ETOT_ * HEADS_) return;
    const int e = id / HEADS_, hh = id - e * HEADS_;
    int s, d; edge_sd(e, edges, s, d);
    float alpha = logits[id] / segsum[d * HEADS_ + hh];
    alpha_out[id] = alpha;
    if (d % N_ == 0) {                     // only ego nodes feed the head
        const int b = d / N_;
        const _Float16* pl = xl + (size_t)s * HC_ + hh * H_;
        for (int c0 = 0; c0 < H_; c0 += 8) {
            h8_t vl = *(const h8_t*)(pl + c0);
#pragma unroll
            for (int j = 0; j < 8; ++j)
                atomicAdd(&node_acc[b * HC_ + hh * H_ + c0 + j],
                          (float)vl[j] * alpha);
        }
    }
}

// ---------------------------------------------------------------- CNN
__global__ void k_conv1(const float* __restrict__ im, const float* __restrict__ w,
                        const float* __restrict__ bias, float* __restrict__ out) {
    int id = blockIdx.x * blockDim.x + threadIdx.x;       // 16*32*16*16
    if (id >= B_ * 32 * 16 * 16) return;
    int x = id & 15, y = (id >> 4) & 15, ch = (id >> 8) & 31, b = id >> 13;
    float acc = bias[ch];
    for (int ci = 0; ci < 4; ++ci)
        for (int ky = 0; ky < 3; ++ky)
            for (int kx = 0; kx < 3; ++kx) {
                int iy = y * 2 + ky, ix = x * 2 + kx;     // SAME: pad_before = 0
                if (iy < 32 && ix < 32)
                    acc += im[((b * 4 + ci) * 32 + iy) * 32 + ix] *
                           w[((ch * 4 + ci) * 3 + ky) * 3 + kx];
            }
    out[id] = fmaxf(acc, 0.f);
}

__global__ void k_conv2(const float* __restrict__ in, const float* __restrict__ w,
                        const float* __restrict__ bias, float* __restrict__ out) {
    int id = blockIdx.x * blockDim.x + threadIdx.x;       // 16*64*8*8
    if (id >= B_ * 64 * 8 * 8) return;
    int x = id & 7, y = (id >> 3) & 7, ch = (id >> 6) & 63, b = id >> 12;
    float acc = bias[ch];
    for (int ci = 0; ci < 32; ++ci)
        for (int ky = 0; ky < 3; ++ky)
            for (int kx = 0; kx < 3; ++kx) {
                int iy = y * 2 + ky, ix = x * 2 + kx;
                if (iy < 16 && ix < 16)
                    acc += in[((b * 32 + ci) * 16 + iy) * 16 + ix] *
                           w[((ch * 32 + ci) * 3 + ky) * 3 + kx];
            }
    out[id] = fmaxf(acc, 0.f);
}

__global__ __launch_bounds__(128) void k_fc_target(
        const float* __restrict__ c2, const float* __restrict__ Wc,
        const float* __restrict__ bc, float* __restrict__ temb) {
    const int b = blockIdx.x, j = threadIdx.x;
    float acc = bc[j];
    const float* row = c2 + (size_t)b * 4096;
    for (int k = 0; k < 4096; ++k) acc += row[k] * Wc[(size_t)k * H_ + j];
    temb[b * H_ + j] = fmaxf(acc, 0.f);
}

// ---------------------------------------------------------------- head
__global__ __launch_bounds__(128) void k_head(
        const float* __restrict__ node_acc, const float* __restrict__ gat_bias,
        const float* __restrict__ temb, const float* __restrict__ phase,
        const float* __restrict__ Wp, const float* __restrict__ bp,
        const float* __restrict__ Wh, const float* __restrict__ bh,
        const float* __restrict__ Wo, const float* __restrict__ bo,
        const float* __restrict__ Wa, const float* __restrict__ ba,
        float* __restrict__ out) {
    __shared__ float fused[264];
    __shared__ float hid[128];
    __shared__ float adv[9];
    const int b = blockIdx.x, j = threadIdx.x;
    float og = 0.f;
    for (int hh = 0; hh < HEADS_; ++hh) og += node_acc[b * HC_ + hh * H_ + j];
    fused[j]       = og * 0.2f + gat_bias[j];             // head mean + bias
    fused[128 + j] = temb[b * H_ + j];
    if (j < 8) {
        float acc = bp[j];
        for (int q = 0; q < 8; ++q) acc += phase[b * 8 + q] * Wp[q * 8 + j];
        fused[256 + j] = fmaxf(acc, 0.f);
    }
    __syncthreads();
    float acc = bh[j];
    for (int k = 0; k < 264; ++k) acc += fused[k] * Wh[k * H_ + j];
    hid[j] = fmaxf(acc, 0.f);
    __syncthreads();
    if (j < 8) {
        float a = ba[j];
        for (int k = 0; k < H_; ++k) a += hid[k] * Wa[k * 8 + j];
        adv[j] = a;
    } else if (j == 8) {
        float v = bo[0];
        for (int k = 0; k < H_; ++k) v += hid[k] * Wo[k];
        adv[8] = v;
    }
    __syncthreads();
    if (j == 0) {
        float mean = 0.f;
        for (int a = 0; a < 8; ++a) mean += adv[a];
        mean *= 0.125f;
        for (int a = 0; a < 8; ++a) out[b * 8 + a] = adv[8] + adv[a] - mean;
    }
}

// ================================================================ launch
extern "C" void kernel_launch(void* const* d_in, const int* in_sizes, int n_in,
                              void* d_out, int out_size, void* d_ws, size_t ws_size,
                              hipStream_t stream) {
    const float* obs      = (const float*)d_in[0];
    const float* phase    = (const float*)d_in[1];
    const float* obs_map  = (const float*)d_in[2];
    const int*   edges    = (const int*)  d_in[3];
    // d_in[4] edges_feature: unused by the reference
    const float* W_nbrs   = (const float*)d_in[5];
    const float* b_nbrs   = (const float*)d_in[6];
    const float* W_ih     = (const float*)d_in[7];
    const float* W_hh     = (const float*)d_in[8];
    const float* b_ih     = (const float*)d_in[9];
    const float* b_hh     = (const float*)d_in[10];
    const float* Wl       = (const float*)d_in[11];
    const float* Wr       = (const float*)d_in[12];
    const float* att_a    = (const float*)d_in[13];
    const float* gat_bias = (const float*)d_in[14];
    const float* c1w      = (const float*)d_in[15];
    const float* c1b      = (const float*)d_in[16];
    const float* c2w      = (const float*)d_in[17];
    const float* c2b      = (const float*)d_in[18];
    const float* Wc       = (const float*)d_in[19];
    const float* bc       = (const float*)d_in[20];
    const float* Wp       = (const float*)d_in[21];
    const float* bp       = (const float*)d_in[22];
    const float* Wh       = (const float*)d_in[23];
    const float* bh       = (const float*)d_in[24];
    const float* Wo       = (const float*)d_in[25];
    const float* bo       = (const float*)d_in[26];
    const float* Wa       = (const float*)d_in[27];
    const float* ba       = (const float*)d_in[28];

    float* outp = (float*)d_out;              // [0,128): Q ; [128,400128): alpha

    // ---- workspace carve (~50 MB)
    char* p = (char*)d_ws;
    auto carve = [&](size_t bytes) { char* r = p; p += (bytes + 255) & ~(size_t)255; return r; };
    _Float16* h16   = (_Float16*)carve((size_t)BN_ * H_ * 2);
    _Float16* Wih16 = (_Float16*)carve((size_t)G3_ * H_ * 2);
    _Float16* Whh16 = (_Float16*)carve((size_t)G3_ * H_ * 2);
    _Float16* WlT   = (_Float16*)carve((size_t)HC_ * H_ * 2);
    _Float16* WrT   = (_Float16*)carve((size_t)HC_ * H_ * 2);
    _Float16* xl    = (_Float16*)carve((size_t)BN_ * HC_ * 2);
    _Float16* xr    = (_Float16*)carve((size_t)BN_ * HC_ * 2);
    float*    logits= (float*)   carve((size_t)ETOT_ * HEADS_ * 4);
    unsigned* segmax= (unsigned*)carve((size_t)BN_ * HEADS_ * 4);
    float*    segsum= (float*)   carve((size_t)BN_ * HEADS_ * 4);
    float*    nacc  = (float*)   carve((size_t)B_ * HC_ * 4);
    float*    c1o   = (float*)   carve((size_t)B_ * 32 * 16 * 16 * 4);
    float*    c2o   = (float*)   carve((size_t)B_ * 64 * 8 * 8 * 4);
    float*    temb  = (float*)   carve((size_t)B_ * H_ * 4);

    // ---- weight conversion + accumulator zeroing
    k_convert<<<(2 * G3_ * H_ + 2 * HC_ * H_ + 255) / 256, 256, 0, stream>>>(
        W_ih, W_hh, Wl, Wr, Wih16, Whh16, WlT, WrT);
    k_zero<<<(BN_ * HEADS_ + 255) / 256, 256, 0, stream>>>((float*)segmax, BN_ * HEADS_);
    k_zero<<<(BN_ * HEADS_ + 255) / 256, 256, 0, stream>>>(segsum, BN_ * HEADS_);
    k_zero<<<(B_ * HC_ + 255) / 256, 256, 0, stream>>>(nacc, B_ * HC_);

    // ---- GRU (WMMA, persistent over T) : LDS = xs + h0 + h1 = 48 KB
    const size_t gru_lds = (size_t)ROWS_ * H_ * 2 * 3;
    k_gru<<<BN_ / ROWS_, 256, gru_lds, stream>>>(
        obs, W_nbrs, b_nbrs, b_ih, b_hh, Wih16, Whh16, h16);

    // ---- GAT projections (WMMA)
    k_gat_gemm<<<BN_ / 16, 256, 0, stream>>>(h16, WlT, WrT, xl, xr);

    // ---- CNN branch (independent, small)
    k_conv1<<<(B_ * 32 * 16 * 16 + 255) / 256, 256, 0, stream>>>(obs_map, c1w, c1b, c1o);
    k_conv2<<<(B_ * 64 * 8 * 8 + 255) / 256, 256, 0, stream>>>(c1o, c2w, c2b, c2o);
    k_fc_target<<<B_, 128, 0, stream>>>(c2o, Wc, bc, temb);

    // ---- edge softmax (alpha is output #2) + ego scatter
    const int eth = ETOT_ * HEADS_;
    k_edge_logits<<<(eth + 255) / 256, 256, 0, stream>>>(xl, xr, att_a, edges, logits, segmax);
    k_edge_exp   <<<(eth + 255) / 256, 256, 0, stream>>>(edges, logits, segmax, segsum);
    k_edge_alpha <<<(eth + 255) / 256, 256, 0, stream>>>(edges, logits, segsum, xl,
                                                         outp + B_ * 8, nacc);

    // ---- fusion + dueling head -> out[0:128)
    k_head<<<B_, 128, 0, stream>>>(nacc, gat_bias, temb, phase, Wp, bp,
                                   Wh, bh, Wo, bo, Wa, ba, outp);
}
